// HGFormerStage_58557584113734
// MI455X (gfx1250) — matrix operand order repaired and checked by reading, not compiled, not run on hardware
//
#include <hip/hip_runtime.h>
#include <hip/hip_bf16.h>

// ---------------- bf16 helpers (raw ushort storage) ----------------
__device__ __forceinline__ unsigned short f2bf(float f) {
  unsigned u = __float_as_uint(f);
  unsigned r = u + 0x7FFFu + ((u >> 16) & 1u);   // round-to-nearest-even
  return (unsigned short)(r >> 16);
}
__device__ __forceinline__ float bf2f(unsigned short h) {
  return __uint_as_float(((unsigned)h) << 16);
}
__device__ __forceinline__ float gelu_exact(float v) {
  return 0.5f * v * (1.0f + erff(v * 0.70710678118654752f));
}

// ---------------- constants ----------------
#define Bb   2
#define Cc   192
#define C4   768
#define Nn   8192          // 8*32*32 tokens per batch
#define Mtot 16384         // B*N
#define NE_  256
#define TOPK_ 32
#define HEADS_ 8
#define DH_  24
#define EPS_ 1e-5f

// ---------------- WMMA GEMM: C[M,Nc] = A[M,K](bf16) * Wt[Nc,K]^T(bf16) + bias ----------------
// 64x64 macro-tile per wave: 4x4 subtiles of 16x16, K-step 32.
// Per K-step: 16 b128 loads feed 16 v_wmma_f32_16x16x32_bf16 (1:1 load:wmma).
typedef __attribute__((ext_vector_type(16))) __bf16 bf16x16;
typedef __attribute__((ext_vector_type(8)))  float  floatx8;
union V16 { bf16x16 v; uint4 q[2]; };

#define MT_ 4
#define NT_ 4

// flags: bit0 = exact GELU, bit1 = residual add into Cf (with optional validMask per row)
__global__ void __launch_bounds__(256) gemm_wmma(
    const unsigned short* __restrict__ A, int lda,
    const unsigned short* __restrict__ Bt, int ldb,
    const float* __restrict__ bias,
    float* __restrict__ Cf, unsigned short* __restrict__ Cbf, int ldc,
    const unsigned int* __restrict__ validMask,
    int M, int Nc, int K, int flags)
{
  int wave = threadIdx.x >> 5;
  int lane = threadIdx.x & 31;
  int tilesN = Nc / 64;
  int totTiles = (M / 64) * tilesN;
  int tile = blockIdx.x * 8 + wave;
  if (tile >= totTiles) return;
  int tm = (tile / tilesN) * 64;
  int tn = (tile % tilesN) * 64;
  int hf  = lane >> 4;        // 0/1
  int l15 = lane & 15;

  const unsigned short* arow[MT_];
  const unsigned short* brow[NT_];
#pragma unroll
  for (int i = 0; i < MT_; ++i)
    arow[i] = A + (size_t)(tm + i * 16 + l15) * lda + hf * 16;   // contiguous 16
#pragma unroll
  for (int j = 0; j < NT_; ++j)
    brow[j] = Bt + (size_t)(tn + j * 16 + l15) * ldb + hf * 16;  // contiguous 16

  floatx8 acc[MT_][NT_];
#pragma unroll
  for (int i = 0; i < MT_; ++i)
#pragma unroll
    for (int j = 0; j < NT_; ++j) acc[i][j] = (floatx8){};

  for (int k0 = 0; k0 < K; k0 += 32) {
    V16 a[MT_], b[NT_];
#pragma unroll
    for (int i = 0; i < MT_; ++i) {
      a[i].q[0] = *(const uint4*)(arow[i] + k0);
      a[i].q[1] = *(const uint4*)(arow[i] + k0 + 8);
    }
#pragma unroll
    for (int j = 0; j < NT_; ++j) {
      b[j].q[0] = *(const uint4*)(brow[j] + k0);
      b[j].q[1] = *(const uint4*)(brow[j] + k0 + 8);
    }
#pragma unroll
    for (int i = 0; i < MT_; ++i)
#pragma unroll
      for (int j = 0; j < NT_; ++j) {
#if defined(__gfx1250__)
        acc[i][j] = __builtin_amdgcn_wmma_f32_16x16x32_bf16(false, a[i].v, false, b[j].v,
                                                            (short)0, acc[i][j], false, false);
#else
        acc[i][j][0] += (float)a[i].q[0].x * 0.f + (float)b[j].q[0].x * 0.f;
#endif
      }
  }

#pragma unroll
  for (int j = 0; j < NT_; ++j) {
    int n = tn + j * 16 + l15;
    float bv = bias ? bias[n] : 0.f;
#pragma unroll
    for (int i = 0; i < MT_; ++i) {
#pragma unroll
      for (int jj = 0; jj < 8; ++jj) {
        int m = tm + i * 16 + jj + hf * 8;
        float v = acc[i][j][jj] + bv;
        if (flags & 1) v = gelu_exact(v);
        size_t off = (size_t)m * ldc + n;
        if (flags & 2) {
          float addv = v;
          if (validMask && !validMask[m]) addv = 0.f;
          if (Cf) Cf[off] += addv;
        } else {
          if (Cf)  Cf[off]  = v;
          if (Cbf) Cbf[off] = f2bf(v);
        }
      }
    }
  }
}

// ---------------- weight prep ----------------
// Wt[n*K + k] = bf16(W[k*N + n])
__global__ void __launch_bounds__(256) transposeW(const float* __restrict__ W,
                                                  unsigned short* __restrict__ Wt,
                                                  int K, int N) {
  int g = blockIdx.x * 256 + threadIdx.x;
  if (g >= K * N) return;
  int n = g / K, k = g % K;
  Wt[g] = f2bf(W[(size_t)k * N + n]);
}

// fold BN into conv weights/bias; conv_w already [oc][ci*8+tap] = [oc][k]
__global__ void __launch_bounds__(256) convPrep(const float* __restrict__ cw,
                                                const float* __restrict__ cb,
                                                const float* __restrict__ gam,
                                                const float* __restrict__ bet,
                                                const float* __restrict__ mean,
                                                const float* __restrict__ var,
                                                unsigned short* __restrict__ Wt,
                                                float* __restrict__ bias) {
  int g = blockIdx.x * 256 + threadIdx.x;
  if (g < Cc * C4) {
    int oc = g / C4;
    float s = rsqrtf(var[oc] + EPS_) * gam[oc];
    Wt[g] = f2bf(cw[g] * s);
  }
  if (g < Cc) {
    float s = rsqrtf(var[g] + EPS_) * gam[g];
    bias[g] = (cb[g] - mean[g]) * s + bet[g];
  }
}

// im2col for 2x2x2 stride-2 conv: Xp[(b*N+n)][ci*8+tap]
__global__ void __launch_bounds__(256) buildPatch(const float* __restrict__ x,
                                                  unsigned short* __restrict__ Xp) {
  long g = (long)blockIdx.x * 256 + threadIdx.x;
  if (g >= (long)Mtot * C4) return;
  int col = (int)(g % C4);
  long m  = g / C4;
  int b = (int)(m >> 13);
  int n = (int)(m & 8191);
  int dp = n >> 10, hp = (n >> 5) & 31, wp = n & 31;
  int ci = col >> 3, tap = col & 7;
  int dz = tap >> 2, dy = (tap >> 1) & 1, dx = tap & 1;
  size_t src = ((((size_t)b * 96 + ci) * 16 + (2 * dp + dz)) * 64 + (2 * hp + dy)) * 64
               + (2 * wp + dx);
  Xp[g] = f2bf(x[src]);
}

// ---------------- LayerNorm: wave per row, bf16 out ----------------
__global__ void __launch_bounds__(256) ln_rows(const float* __restrict__ xf,
                                               const float* __restrict__ g,
                                               const float* __restrict__ b,
                                               unsigned short* __restrict__ out, int M) {
  int row = blockIdx.x * 8 + (threadIdx.x >> 5);
  if (row >= M) return;
  int lane = threadIdx.x & 31;
  const float* x = xf + (size_t)row * Cc;
  float v[6];
  float s = 0.f;
#pragma unroll
  for (int t = 0; t < 6; ++t) { v[t] = x[lane + 32 * t]; s += v[t]; }
  for (int o = 16; o > 0; o >>= 1) s += __shfl_xor(s, o);
  float mean = s * (1.f / 192.f);
  float var = 0.f;
#pragma unroll
  for (int t = 0; t < 6; ++t) { float d = v[t] - mean; var += d * d; }
  for (int o = 16; o > 0; o >>= 1) var += __shfl_xor(var, o);
  float rs = rsqrtf(var * (1.f / 192.f) + EPS_);
#pragma unroll
  for (int t = 0; t < 6; ++t) {
    int c = lane + 32 * t;
    out[(size_t)row * Cc + c] = f2bf((v[t] - mean) * rs * g[c] + b[c]);
  }
}

// ---------------- hypergraph: distances + top-32 + membership bitmask ----------------
__global__ void __launch_bounds__(256) hypergraph_kernel(const float* __restrict__ xf,
                                                         int* __restrict__ idxOut,
                                                         unsigned int* __restrict__ Hbits) {
  __shared__ float dist[Nn];
  __shared__ float cf[Cc];
  __shared__ float rv[256];
  __shared__ int   ri[256];
  int e = blockIdx.x & 255;
  int b = blockIdx.x >> 8;
  int tid = threadIdx.x;
  int ce = (int)rintf((float)e * (8191.0f / 255.0f));
  if (tid < Cc) cf[tid] = xf[((size_t)b * Nn + ce) * Cc + tid];
  __syncthreads();
  float cd = 2.f * (float)(ce >> 10), chh = (float)((ce >> 5) & 31), cw = (float)(ce & 31);
  for (int n = tid; n < Nn; n += 256) {
    const float* xr = xf + ((size_t)b * Nn + n) * Cc;
    float dd = 2.f * (float)(n >> 10) - cd;
    float dh = (float)((n >> 5) & 31) - chh;
    float dw = (float)(n & 31) - cw;
    float s = dd * dd + dh * dh + dw * dw;
    for (int c = 0; c < Cc; ++c) { float d = cf[c] - xr[c]; s += d * d; }
    dist[n] = s;
  }
  __syncthreads();
  for (int it = 0; it < TOPK_; ++it) {
    float best = 3.4e38f; int bi = Nn;
    for (int n = tid; n < Nn; n += 256) {
      float v = dist[n];
      if (v < best) { best = v; bi = n; }
    }
    rv[tid] = best; ri[tid] = bi;
    __syncthreads();
    for (int s = 128; s > 0; s >>= 1) {
      if (tid < s) {
        if (rv[tid + s] < rv[tid] || (rv[tid + s] == rv[tid] && ri[tid + s] < ri[tid])) {
          rv[tid] = rv[tid + s]; ri[tid] = ri[tid + s];
        }
      }
      __syncthreads();
    }
    if (tid == 0) {
      int node = ri[0];
      idxOut[(b * NE_ + e) * TOPK_ + it] = node;
      dist[node] = 3.4e38f;
      atomicOr(&Hbits[((size_t)b * Nn + node) * 8 + (e >> 5)], 1u << (e & 31));
    }
    __syncthreads();
  }
}

// ---------------- he0 = mean of member features (deg == 32 always) ----------------
__global__ void __launch_bounds__(256) edgeMean(const unsigned short* __restrict__ xn,
                                                const int* __restrict__ idx,
                                                unsigned short* __restrict__ he0) {
  int g = blockIdx.x * 256 + threadIdx.x;
  if (g >= Bb * NE_ * Cc) return;
  int c = g % Cc;
  int be = g / Cc;
  int b = be >> 8;
  float s = 0.f;
  for (int k = 0; k < TOPK_; ++k) {
    int node = idx[be * TOPK_ + k];
    s += bf2f(xn[((size_t)b * Nn + node) * Cc + c]);
  }
  he0[g] = f2bf(s * (1.f / 32.f));
}

// ---------------- attention 1: edges attend over their 32 members ----------------
__global__ void __launch_bounds__(256) attn1_kernel(const float* __restrict__ q1,
                                                    const unsigned short* __restrict__ k1,
                                                    const unsigned short* __restrict__ v1,
                                                    const int* __restrict__ idx,
                                                    unsigned short* __restrict__ ho1) {
  __shared__ int   sidx[TOPK_];
  __shared__ float sq[Cc];
  __shared__ float sattn[HEADS_][TOPK_];
  int be = blockIdx.x;              // b*256 + e
  int b  = be >> 8;
  int tid = threadIdx.x;
  if (tid < TOPK_) sidx[tid] = idx[be * TOPK_ + tid];
  if (tid < Cc)    sq[tid]   = q1[(size_t)be * Cc + tid];
  __syncthreads();
  int h = tid >> 5, m = tid & 31;
  size_t kb = ((size_t)b * Nn + sidx[m]) * Cc + h * DH_;
  float s = 0.f;
  for (int d = 0; d < DH_; ++d) s += sq[h * DH_ + d] * bf2f(k1[kb + d]);
  s *= 0.20412414523193151f;        // 1/sqrt(24)
  float mx = s;
  for (int o = 16; o > 0; o >>= 1) mx = fmaxf(mx, __shfl_xor(mx, o));
  float p = __expf(s - mx);
  float sum = p;
  for (int o = 16; o > 0; o >>= 1) sum += __shfl_xor(sum, o);
  sattn[h][m] = p / sum;
  __syncthreads();
  if (tid < Cc) {
    int c = tid, hh = c / DH_;
    float o = 0.f;
    for (int mm = 0; mm < TOPK_; ++mm)
      o += sattn[hh][mm] * bf2f(v1[((size_t)b * Nn + sidx[mm]) * Cc + c]);
    ho1[(size_t)be * Cc + c] = f2bf(o);
  }
}

// ---------------- attention 2: wave per node, online softmax over member edges ----------------
__global__ void __launch_bounds__(256) attn2_kernel(const float* __restrict__ q2,
                                                    const float* __restrict__ k2,
                                                    const float* __restrict__ v2,
                                                    const unsigned int* __restrict__ Hbits,
                                                    unsigned short* __restrict__ ho2,
                                                    unsigned int* __restrict__ valid) {
  __shared__ float sq[8][Cc];
  int w = threadIdx.x >> 5, lane = threadIdx.x & 31;
  int row = blockIdx.x * 8 + w;     // b*N + n
  int b = row >> 13;
#pragma unroll
  for (int t = 0; t < 6; ++t) sq[w][lane + 32 * t] = q2[(size_t)row * Cc + lane + 32 * t];
  __syncthreads();
  const float scale = 0.20412414523193151f;
  float mx[HEADS_], den[HEADS_], acc[6];
  int headOf[6];
#pragma unroll
  for (int h = 0; h < HEADS_; ++h) { mx[h] = -3.0e38f; den[h] = 0.f; }
#pragma unroll
  for (int t = 0; t < 6; ++t) { acc[t] = 0.f; headOf[t] = (lane + 32 * t) / DH_; }
  int cnt = 0;
  for (int w8 = 0; w8 < 8; ++w8) {
    unsigned bits = Hbits[(size_t)row * 8 + w8];
    while (bits) {
      int eb = __ffs(bits) - 1;
      bits &= bits - 1;
      int e = w8 * 32 + eb;
      const float* krow = k2 + (size_t)(b * NE_ + e) * Cc;
      const float* vrow = v2 + (size_t)(b * NE_ + e) * Cc;
      float sh[HEADS_];
#pragma unroll
      for (int h = 0; h < HEADS_; ++h) {
        float p = (lane < DH_) ? sq[w][h * DH_ + lane] * krow[h * DH_ + lane] : 0.f;
        for (int o = 16; o > 0; o >>= 1) p += __shfl_xor(p, o);
        sh[h] = p * scale;
      }
      float corr[HEADS_], pe[HEADS_];
#pragma unroll
      for (int h = 0; h < HEADS_; ++h) {
        float nm = fmaxf(mx[h], sh[h]);
        corr[h] = __expf(mx[h] - nm);
        pe[h]   = __expf(sh[h] - nm);
        den[h]  = den[h] * corr[h] + pe[h];
        mx[h]   = nm;
      }
#pragma unroll
      for (int t = 0; t < 6; ++t) {
        int c = lane + 32 * t;
        int h = headOf[t];
        acc[t] = acc[t] * corr[h] + pe[h] * vrow[c];
      }
      ++cnt;
    }
  }
  if (lane == 0) valid[row] = (cnt > 0) ? 1u : 0u;
#pragma unroll
  for (int t = 0; t < 6; ++t) {
    float o = (cnt > 0) ? acc[t] / den[headOf[t]] : 0.f;
    ho2[(size_t)row * Cc + lane + 32 * t] = f2bf(o);
  }
}

// ---------------- final transpose (B,N,C) -> (B,C,D,H,W) ----------------
__global__ void __launch_bounds__(256) writeOut(const float* __restrict__ xf,
                                                float* __restrict__ out) {
  int g = blockIdx.x * 256 + threadIdx.x;
  if (g >= Bb * Cc * Nn) return;
  int n = g & 8191;
  int c = (g >> 13) % Cc;
  int b = g / (Cc * Nn);
  out[g] = xf[((size_t)b * Nn + n) * Cc + c];
}

// ---------------- host orchestration ----------------
extern "C" void kernel_launch(void* const* d_in, const int* in_sizes, int n_in,
                              void* d_out, int out_size, void* d_ws, size_t ws_size,
                              hipStream_t stream) {
  const float* x       = (const float*)d_in[0];
  const float* conv_w  = (const float*)d_in[1];
  const float* conv_b  = (const float*)d_in[2];
  const float* bn_g    = (const float*)d_in[3];
  const float* bn_b    = (const float*)d_in[4];
  const float* bn_m    = (const float*)d_in[5];
  const float* bn_v    = (const float*)d_in[6];
  const float* ln_g    = (const float*)d_in[7];
  const float* ln_b    = (const float*)d_in[8];
  const float* attn_w  = (const float*)d_in[9];
  const float* attn_b  = (const float*)d_in[10];
  const float* mlp_w1  = (const float*)d_in[11];
  const float* mlp_b1  = (const float*)d_in[12];
  const float* mlp_w2  = (const float*)d_in[13];
  const float* mlp_b2  = (const float*)d_in[14];

  // bump allocator over workspace
  size_t off = 0;
  auto alloc = [&](size_t bytes) -> void* {
    void* p = (char*)d_ws + off;
    off += (bytes + 255) & ~(size_t)255;
    return p;
  };
  unsigned short* convWt  = (unsigned short*)alloc((size_t)Cc * C4 * 2);
  float*          convBias= (float*)alloc((size_t)Cc * 4);
  unsigned short* attnWt  = (unsigned short*)alloc((size_t)16 * Cc * Cc * 2);
  unsigned short* mlpW1t  = (unsigned short*)alloc((size_t)2 * C4 * Cc * 2);
  unsigned short* mlpW2t  = (unsigned short*)alloc((size_t)2 * Cc * C4 * 2);
  unsigned short* bigbuf  = (unsigned short*)alloc((size_t)Mtot * C4 * 2); // Xpatch, then MLP hidden
  float*          xf      = (float*)alloc((size_t)Mtot * Cc * 4);
  unsigned short* xn      = (unsigned short*)alloc((size_t)Mtot * Cc * 2);
  unsigned short* k1      = (unsigned short*)alloc((size_t)Mtot * Cc * 2);
  unsigned short* v1      = (unsigned short*)alloc((size_t)Mtot * Cc * 2);
  float*          q2      = (float*)alloc((size_t)Mtot * Cc * 4);
  unsigned short* ho2     = (unsigned short*)alloc((size_t)Mtot * Cc * 2);
  unsigned int*   valid   = (unsigned int*)alloc((size_t)Mtot * 4);
  int*            idx     = (int*)alloc((size_t)Bb * NE_ * TOPK_ * 4);
  unsigned int*   Hbits   = (unsigned int*)alloc((size_t)Bb * Nn * 8 * 4);
  unsigned short* he0     = (unsigned short*)alloc((size_t)Bb * NE_ * Cc * 2);
  float*          q1      = (float*)alloc((size_t)Bb * NE_ * Cc * 4);
  unsigned short* ho1     = (unsigned short*)alloc((size_t)Bb * NE_ * Cc * 2);
  unsigned short* heb     = (unsigned short*)alloc((size_t)Bb * NE_ * Cc * 2);
  float*          k2      = (float*)alloc((size_t)Bb * NE_ * Cc * 4);
  float*          v2      = (float*)alloc((size_t)Bb * NE_ * Cc * 4);
  (void)ws_size; (void)in_sizes; (void)n_in; (void)out_size;

  auto gemm = [&](const unsigned short* A, int lda, const unsigned short* Bt, int ldb,
                  const float* bias, float* Cf, unsigned short* Cbf, int ldc,
                  const unsigned int* vmask, int M, int Nc, int K, int flags) {
    int tiles = (M / 64) * (Nc / 64);
    gemm_wmma<<<(tiles + 7) / 8, 256, 0, stream>>>(A, lda, Bt, ldb, bias, Cf, Cbf, ldc,
                                                   vmask, M, Nc, K, flags);
  };

  // ---- weight prep ----
  for (int t = 0; t < 16; ++t)
    transposeW<<<(Cc * Cc + 255) / 256, 256, 0, stream>>>(attn_w + (size_t)t * Cc * Cc,
                                                          attnWt + (size_t)t * Cc * Cc, Cc, Cc);
  for (int i = 0; i < 2; ++i) {
    transposeW<<<(Cc * C4 + 255) / 256, 256, 0, stream>>>(mlp_w1 + (size_t)i * Cc * C4,
                                                          mlpW1t + (size_t)i * C4 * Cc, Cc, C4);
    transposeW<<<(Cc * C4 + 255) / 256, 256, 0, stream>>>(mlp_w2 + (size_t)i * C4 * Cc,
                                                          mlpW2t + (size_t)i * Cc * C4, C4, Cc);
  }
  convPrep<<<(Cc * C4 + 255) / 256, 256, 0, stream>>>(conv_w, conv_b, bn_g, bn_b, bn_m, bn_v,
                                                      convWt, convBias);

  // ---- conv as GEMM (+BN folded, exact GELU) ----
  buildPatch<<<(Mtot * C4) / 256, 256, 0, stream>>>(x, bigbuf);
  gemm(bigbuf, C4, convWt, C4, convBias, xf, nullptr, Cc, nullptr, Mtot, Cc, C4, /*gelu*/1);

  // ---- hypergraph ----
  hipMemsetAsync(Hbits, 0, (size_t)Bb * Nn * 8 * 4, stream);
  hypergraph_kernel<<<Bb * NE_, 256, 0, stream>>>(xf, idx, Hbits);

  // ---- transformer layers ----
  for (int i = 0; i < 2; ++i) {
    const unsigned short* WT0 = attnWt + (size_t)((i * 2 + 0) * 4) * Cc * Cc;
    const unsigned short* WT1 = attnWt + (size_t)((i * 2 + 1) * 4) * Cc * Cc;
    const float* BA0 = attn_b + (size_t)((i * 2 + 0) * 4) * Cc;
    const float* BA1 = attn_b + (size_t)((i * 2 + 1) * 4) * Cc;

    // MHA #1: edges <- member nodes
    ln_rows<<<Mtot / 8, 256, 0, stream>>>(xf, ln_g + (i * 3 + 0) * Cc, ln_b + (i * 3 + 0) * Cc,
                                          xn, Mtot);
    edgeMean<<<(Bb * NE_ * Cc + 255) / 256, 256, 0, stream>>>(xn, idx, he0);
    gemm(he0, Cc, WT0 + 0 * Cc * Cc, Cc, BA0 + 0 * Cc, q1, nullptr, Cc, nullptr,
         Bb * NE_, Cc, Cc, 0);
    gemm(xn, Cc, WT0 + 1 * Cc * Cc, Cc, BA0 + 1 * Cc, nullptr, k1, Cc, nullptr,
         Mtot, Cc, Cc, 0);
    gemm(xn, Cc, WT0 + 2 * Cc * Cc, Cc, BA0 + 2 * Cc, nullptr, v1, Cc, nullptr,
         Mtot, Cc, Cc, 0);
    attn1_kernel<<<Bb * NE_, 256, 0, stream>>>(q1, k1, v1, idx, ho1);
    gemm(ho1, Cc, WT0 + 3 * Cc * Cc, Cc, BA0 + 3 * Cc, nullptr, heb, Cc, nullptr,
         Bb * NE_, Cc, Cc, 0);

    // MHA #2: nodes <- member edges (residual into xf, masked by valid)
    ln_rows<<<Mtot / 8, 256, 0, stream>>>(xf, ln_g + (i * 3 + 1) * Cc, ln_b + (i * 3 + 1) * Cc,
                                          xn, Mtot);
    gemm(xn, Cc, WT1 + 0 * Cc * Cc, Cc, BA1 + 0 * Cc, q2, nullptr, Cc, nullptr,
         Mtot, Cc, Cc, 0);
    gemm(heb, Cc, WT1 + 1 * Cc * Cc, Cc, BA1 + 1 * Cc, k2, nullptr, Cc, nullptr,
         Bb * NE_, Cc, Cc, 0);
    gemm(heb, Cc, WT1 + 2 * Cc * Cc, Cc, BA1 + 2 * Cc, v2, nullptr, Cc, nullptr,
         Bb * NE_, Cc, Cc, 0);
    attn2_kernel<<<Mtot / 8, 256, 0, stream>>>(q2, k2, v2, Hbits, ho2, valid);
    gemm(ho2, Cc, WT1 + 3 * Cc * Cc, Cc, BA1 + 3 * Cc, xf, nullptr, Cc, valid,
         Mtot, Cc, Cc, /*residual*/2);

    // MLP (residual into xf)
    ln_rows<<<Mtot / 8, 256, 0, stream>>>(xf, ln_g + (i * 3 + 2) * Cc, ln_b + (i * 3 + 2) * Cc,
                                          xn, Mtot);
    gemm(xn, Cc, mlpW1t + (size_t)i * C4 * Cc, Cc, mlp_b1 + i * C4, nullptr, bigbuf, C4,
         nullptr, Mtot, C4, Cc, /*gelu*/1);
    gemm(bigbuf, C4, mlpW2t + (size_t)i * Cc * C4, C4, mlp_b2 + i * Cc, xf, nullptr, Cc,
         nullptr, Mtot, Cc, C4, /*residual*/2);
  }

  // ---- output ----
  writeOut<<<(Bb * Cc * Nn) / 256, 256, 0, stream>>>(xf, (float*)d_out);
}
